// Attention_5162550690111
// MI455X (gfx1250) — matrix operand order repaired and checked by reading, not compiled, not run on hardware
//
#include <hip/hip_runtime.h>
#include <hip/hip_bf16.h>

typedef __attribute__((ext_vector_type(16))) _Float16 v16h;
typedef __attribute__((ext_vector_type(8)))  float    v8f;

union frag_u { v16h v; float4 f4[2]; };

#define DIMX   1024
#define HEADS  16
#define DH     64
#define INNER  1024
#define SEQ    2048
#define BATCH  2
#define MTOK   (BATCH * SEQ)       // 4096
#define NEG_BIG (-3.0e38f)

// Async copy: global -> LDS, 16 bytes per lane, tracked with ASYNCcnt.
__device__ __forceinline__ void async_copy_b128(void* lds_dst, const void* gsrc)
{
    unsigned lds_off = (unsigned)(size_t)lds_dst;   // addrspace(3) offset
    asm volatile("global_load_async_to_lds_b128 %0, %1, off"
                 :: "v"(lds_off), "v"(gsrc) : "memory");
}
__device__ __forceinline__ void wait_async_all()
{
    asm volatile("s_wait_asynccnt 0x0" ::: "memory");
}

// ---------------------------------------------------------------------------
// Kernel 1: LayerNorm  (f32 in -> f16 out)
// ---------------------------------------------------------------------------
__global__ void __launch_bounds__(256) ln_kernel(
    const float* __restrict__ x, const float* __restrict__ g,
    const float* __restrict__ beta, _Float16* __restrict__ xn)
{
    const int row = blockIdx.x;
    const float* xr = x + (size_t)row * DIMX;
    __shared__ float red[256];

    float vals[4];
    float s = 0.f;
    #pragma unroll
    for (int i = 0; i < 4; ++i) { vals[i] = xr[threadIdx.x + i * 256]; s += vals[i]; }

    red[threadIdx.x] = s; __syncthreads();
    for (int off = 128; off > 0; off >>= 1) {
        if ((int)threadIdx.x < off) red[threadIdx.x] += red[threadIdx.x + off];
        __syncthreads();
    }
    const float mu = red[0] * (1.0f / DIMX);
    __syncthreads();

    float vs = 0.f;
    #pragma unroll
    for (int i = 0; i < 4; ++i) { float d = vals[i] - mu; vs += d * d; }
    red[threadIdx.x] = vs; __syncthreads();
    for (int off = 128; off > 0; off >>= 1) {
        if ((int)threadIdx.x < off) red[threadIdx.x] += red[threadIdx.x + off];
        __syncthreads();
    }
    const float rstd = rsqrtf(red[0] * (1.0f / DIMX) + 1e-5f);

    #pragma unroll
    for (int i = 0; i < 4; ++i) {
        int c = threadIdx.x + i * 256;
        float y = (vals[i] - mu) * rstd * g[c] + beta[c];
        xn[(size_t)row * DIMX + c] = (_Float16)y;
    }
}

// ---------------------------------------------------------------------------
// Kernel 2/4: WMMA GEMM  C[M,N] = A_f16[M,K] * B_f32[K,N]
//   Double-buffered pipeline: async-DMA A tile + register-staged B tile for
//   k-step i+1 issued before the 8 WMMAs of k-step i; one barrier/iteration.
//   MODE 0: write f32 to Cout;  MODE 1: split columns into q/k/v f16 buffers.
// Tiles: 128x128x32, 256 threads (8 waves), wave grid 4x2, wave tile 32x64.
// ---------------------------------------------------------------------------
#define BM 128
#define BN 128
#define BK 32

template <int N, int K, int MODE>
__global__ void __launch_bounds__(256) gemm_kernel(
    const _Float16* __restrict__ A, const float* __restrict__ Bm,
    _Float16* __restrict__ qb, _Float16* __restrict__ kb,
    _Float16* __restrict__ vb, float* __restrict__ Cout)
{
    __shared__ _Float16 As[2][BM][BK + 8];    // double-buffered (m,k)
    __shared__ _Float16 Bt[2][BN][BK + 8];    // double-buffered transposed (n,k)

    const int tid  = threadIdx.x;
    const int wave = tid >> 5;
    const int lane = tid & 31;
    const int l16  = lane & 15;
    const int hf   = lane >> 4;
    const int m0   = (wave >> 1) * 32;
    const int n0   = (wave & 1) * 64;
    const int tileM = blockIdx.x * BM;
    const int tileN = blockIdx.y * BN;

    v8f acc[2][4];
    v8f zero = {};
    #pragma unroll
    for (int i = 0; i < 2; ++i)
        #pragma unroll
        for (int j = 0; j < 4; ++j) acc[i][j] = zero;

    // staging coordinates (fixed per thread)
    const int ar = tid >> 1;             // A: row 0..127
    const int ac = (tid & 1) * 16;       // A: col 0 or 16
    const int bn = tid >> 1;             // B: column 0..127
    const int bkh = tid & 1;             // B: k-half (0 or 1)

    const _Float16* asrc = A + (size_t)(tileM + ar) * K + ac;
    const float*    bsrc = Bm + (size_t)(bkh * 16) * N + tileN + bn;

    // ---- prologue: stage k-step 0 into buffer 0 ----
    async_copy_b128(&As[0][ar][ac],     asrc);
    async_copy_b128(&As[0][ar][ac + 8], asrc + 8);
    {
        alignas(16) _Float16 tmp[16];
        #pragma unroll
        for (int j = 0; j < 16; ++j)
            tmp[j] = (_Float16)bsrc[(size_t)j * N];
        *(float4*)(&Bt[0][bn][bkh * 16])     = *(const float4*)(&tmp[0]);
        *(float4*)(&Bt[0][bn][bkh * 16 + 8]) = *(const float4*)(&tmp[8]);
    }
    wait_async_all();
    __syncthreads();

    int buf = 0;
    for (int k0 = 0; k0 < K; k0 += BK, buf ^= 1) {
        const bool has_next = (k0 + BK < K);
        float bregs[16];

        // (1) issue next tile: async A copy + B global loads into registers
        if (has_next) {
            async_copy_b128(&As[buf ^ 1][ar][ac],     asrc + k0 + BK);
            async_copy_b128(&As[buf ^ 1][ar][ac + 8], asrc + k0 + BK + 8);
            const float* src = bsrc + (size_t)(k0 + BK) * N;
            #pragma unroll
            for (int j = 0; j < 16; ++j)
                bregs[j] = src[(size_t)j * N];       // immediate offsets
        }

        // (2) compute on current buffer
        v16h af[2];
        #pragma unroll
        for (int i = 0; i < 2; ++i) {
            const _Float16* ap = &As[buf][m0 + i * 16 + l16][0];
            frag_u u;
            u.f4[0] = *(const float4*)(ap + hf * 8);
            u.f4[1] = *(const float4*)(ap + 16 + hf * 8);
            af[i] = u.v;
        }
        #pragma unroll
        for (int jf = 0; jf < 4; ++jf) {
            const _Float16* bp = &Bt[buf][n0 + jf * 16 + l16][0];
            frag_u u;
            u.f4[0] = *(const float4*)(bp + hf * 8);
            u.f4[1] = *(const float4*)(bp + 16 + hf * 8);
            acc[0][jf] = __builtin_amdgcn_wmma_f32_16x16x32_f16(
                false, af[0], false, u.v, (short)0, acc[0][jf], false, false);
            acc[1][jf] = __builtin_amdgcn_wmma_f32_16x16x32_f16(
                false, af[1], false, u.v, (short)0, acc[1][jf], false, false);
        }

        // (3) convert + store next B tile into the other buffer
        if (has_next) {
            alignas(16) _Float16 tmp[16];
            #pragma unroll
            for (int j = 0; j < 16; ++j)
                tmp[j] = (_Float16)bregs[j];
            *(float4*)(&Bt[buf ^ 1][bn][bkh * 16])     = *(const float4*)(&tmp[0]);
            *(float4*)(&Bt[buf ^ 1][bn][bkh * 16 + 8]) = *(const float4*)(&tmp[8]);
        }

        // (4) make next buffer visible
        wait_async_all();
        __syncthreads();
    }

    // ---- epilogue ----
    #pragma unroll
    for (int i = 0; i < 2; ++i) {
        #pragma unroll
        for (int jf = 0; jf < 4; ++jf) {
            int col = tileN + n0 + jf * 16 + l16;
            #pragma unroll
            for (int r = 0; r < 8; ++r) {
                int row = tileM + m0 + i * 16 + r + hf * 8;
                float val = acc[i][jf][r];
                if (MODE == 0) {
                    Cout[(size_t)row * N + col] = val;
                } else {
                    int which = col >> 10;          // 0:q 1:k 2:v
                    int cc = col & 1023;
                    int h  = cc >> 6;
                    int d  = cc & 63;
                    int b  = row >> 11;
                    int n  = row & (SEQ - 1);
                    size_t idx = ((size_t)(b * HEADS + h) * SEQ + n) * DH + d;
                    _Float16 hv = (_Float16)val;
                    if (which == 0)      qb[idx] = hv;
                    else if (which == 1) kb[idx] = hv;
                    else                 vb[idx] = hv;
                }
            }
        }
    }
}

// ---------------------------------------------------------------------------
// Kernel 3: flash attention, double-buffered K/V tiles.
// Block = 128 threads (4 waves), each wave owns 16 query rows; block covers
// (b,h, 64 q rows). K loop in tiles of 32. K tile async-DMA'd row-major
// [k][d]; V tile register-staged TRANSPOSED [d][k].
// ---------------------------------------------------------------------------
__global__ void __launch_bounds__(128) attn_kernel(
    const _Float16* __restrict__ qb, const _Float16* __restrict__ kb,
    const _Float16* __restrict__ vb, const int* __restrict__ mask,
    _Float16* __restrict__ ao)
{
    __shared__ _Float16 Ks[2][32][DH + 8];   // [k][d], double-buffered
    __shared__ _Float16 Vt[2][DH][40];       // [d][k] transposed, double-buffered
    __shared__ _Float16 Ps[4][16][40];       // per-wave P tile [q][k]

    const int blk  = blockIdx.x;
    const int qt   = blk & 31;            // SEQ/64 = 32 q-tiles
    const int bh   = blk >> 5;            // b*HEADS + h
    const int tid  = threadIdx.x;
    const int wave = tid >> 5;
    const int lane = tid & 31;
    const int l16  = lane & 15;
    const int hf   = lane >> 4;
    const int q0   = qt * 64 + wave * 16;
    const int b    = bh >> 4;
    const int h    = bh & 15;

    const _Float16* qp = qb + (size_t)bh * SEQ * DH;
    const _Float16* kp = kb + (size_t)bh * SEQ * DH;
    const _Float16* vp = vb + (size_t)bh * SEQ * DH;
    const int* mrow = mask + (size_t)bh * SEQ * SEQ;

    // Q fragments for this wave's 16 rows (two 16x32 A frags over d)
    v16h aq[2];
    {
        const _Float16* qr = qp + (size_t)(q0 + l16) * DH;
        #pragma unroll
        for (int i = 0; i < 2; ++i) {
            frag_u u;
            u.f4[0] = *(const float4*)(qr + i * 32 + hf * 8);
            u.f4[1] = *(const float4*)(qr + i * 32 + 16 + hf * 8);
            aq[i] = u.v;
        }
    }

    v8f oacc[4];
    v8f zero = {};
    #pragma unroll
    for (int i = 0; i < 4; ++i) oacc[i] = zero;
    float rowm[8], rowl[8];
    #pragma unroll
    for (int r = 0; r < 8; ++r) { rowm[r] = NEG_BIG; rowl[r] = 0.f; }

    // staging coords
    const int kr = tid >> 2;             // K: row 0..31
    const int kc = (tid & 3) * 16;       // K: col 0/16/32/48
    const int vd = tid >> 1;             // V: d 0..63
    const int vkh = tid & 1;             // V: k-half

    const _Float16* kstage = kp + (size_t)kr * DH + kc;
    const _Float16* vstage = vp + (size_t)(vkh * 16) * DH + vd;

    // ---- prologue: stage tile 0 into buffer 0 ----
    async_copy_b128(&Ks[0][kr][kc],     kstage);
    async_copy_b128(&Ks[0][kr][kc + 8], kstage + 8);
    {
        alignas(16) _Float16 tmp[16];
        #pragma unroll
        for (int j = 0; j < 16; ++j)
            tmp[j] = vstage[(size_t)j * DH];
        *(float4*)(&Vt[0][vd][vkh * 16])     = *(const float4*)(&tmp[0]);
        *(float4*)(&Vt[0][vd][vkh * 16 + 8]) = *(const float4*)(&tmp[8]);
    }
    wait_async_all();
    __syncthreads();

    int buf = 0;
    for (int kt = 0; kt < SEQ; kt += 32, buf ^= 1) {
        const bool has_next = (kt + 32 < SEQ);
        _Float16 vregs[16];

        // (1) issue next tile: async K copy + V global loads into registers
        if (has_next) {
            const size_t nxt = (size_t)(kt + 32) * DH;
            async_copy_b128(&Ks[buf ^ 1][kr][kc],     kstage + nxt);
            async_copy_b128(&Ks[buf ^ 1][kr][kc + 8], kstage + nxt + 8);
            const _Float16* vs = vstage + nxt;
            #pragma unroll
            for (int j = 0; j < 16; ++j)
                vregs[j] = vs[(size_t)j * DH];        // immediate offsets
        }

        // (2a) S = Q * K^T  (two 16-col fragments, accumulate over d)
        v8f sc[2];
        #pragma unroll
        for (int c = 0; c < 2; ++c) {
            v8f z = zero;
            #pragma unroll
            for (int dc = 0; dc < 2; ++dc) {
                const _Float16* bp = &Ks[buf][c * 16 + l16][dc * 32];
                frag_u u;
                u.f4[0] = *(const float4*)(bp + hf * 8);
                u.f4[1] = *(const float4*)(bp + 16 + hf * 8);
                z = __builtin_amdgcn_wmma_f32_16x16x32_f16(
                    false, aq[dc], false, u.v, (short)0, z, false, false);
            }
            sc[c] = z;
        }

        // (2b) scale + mask
        float s[2][8];
        #pragma unroll
        for (int c = 0; c < 2; ++c)
            #pragma unroll
            for (int r = 0; r < 8; ++r) {
                int q    = q0 + r + hf * 8;
                int kcol = kt + c * 16 + l16;
                float v = sc[c][r] * 0.125f;   // 1/sqrt(64)
                s[c][r] = (mrow[(size_t)q * SEQ + kcol] != 0) ? NEG_BIG : v;
            }

        // (2c) online softmax update
        float corr[8];
        #pragma unroll
        for (int r = 0; r < 8; ++r) {
            float m = fmaxf(s[0][r], s[1][r]);
            #pragma unroll
            for (int off = 1; off < 16; off <<= 1)
                m = fmaxf(m, __shfl_xor(m, off, 32));
            float nm = fmaxf(rowm[r], m);
            corr[r] = __expf(rowm[r] - nm);
            float sum = 0.f;
            #pragma unroll
            for (int c = 0; c < 2; ++c) {
                float p = __expf(s[c][r] - nm);
                s[c][r] = p;
                sum += p;
            }
            #pragma unroll
            for (int off = 1; off < 16; off <<= 1)
                sum += __shfl_xor(sum, off, 32);
            rowl[r] = rowl[r] * corr[r] + sum;
            rowm[r] = nm;
        }

        // (2d) restage P (C layout -> A layout via per-wave LDS)
        #pragma unroll
        for (int c = 0; c < 2; ++c)
            #pragma unroll
            for (int r = 0; r < 8; ++r)
                Ps[wave][r + hf * 8][c * 16 + l16] = (_Float16)s[c][r];
        #pragma unroll
        for (int nf = 0; nf < 4; ++nf)
            #pragma unroll
            for (int r = 0; r < 8; ++r)
                oacc[nf][r] *= corr[r];

        v16h ap;
        {
            const _Float16* pp = &Ps[wave][l16][0];
            frag_u u;
            u.f4[0] = *(const float4*)(pp + hf * 8);
            u.f4[1] = *(const float4*)(pp + 16 + hf * 8);
            ap = u.v;
        }

        // (2e) O += P * V
        #pragma unroll
        for (int nf = 0; nf < 4; ++nf) {
            const _Float16* bp = &Vt[buf][nf * 16 + l16][0];
            frag_u u;
            u.f4[0] = *(const float4*)(bp + hf * 8);
            u.f4[1] = *(const float4*)(bp + 16 + hf * 8);
            oacc[nf] = __builtin_amdgcn_wmma_f32_16x16x32_f16(
                false, ap, false, u.v, (short)0, oacc[nf], false, false);
        }

        // (3) store next V tile transposed into the other buffer
        if (has_next) {
            alignas(16) _Float16 tmp[16];
            #pragma unroll
            for (int j = 0; j < 16; ++j)
                tmp[j] = vregs[j];
            *(float4*)(&Vt[buf ^ 1][vd][vkh * 16])     = *(const float4*)(&tmp[0]);
            *(float4*)(&Vt[buf ^ 1][vd][vkh * 16 + 8]) = *(const float4*)(&tmp[8]);
        }

        // (4) make next buffer visible
        wait_async_all();
        __syncthreads();
    }

    // ---- finalize: O /= l, write f16 to [b, n, h*64+d] ----
    #pragma unroll
    for (int r = 0; r < 8; ++r) rowl[r] = 1.0f / rowl[r];
    #pragma unroll
    for (int nf = 0; nf < 4; ++nf)
        #pragma unroll
        for (int r = 0; r < 8; ++r) {
            int qq = q0 + r + hf * 8;
            int d  = nf * 16 + l16;
            float val = oacc[nf][r] * rowl[r];
            ao[((size_t)(b * SEQ + qq)) * INNER + h * DH + d] = (_Float16)val;
        }
}

// ---------------------------------------------------------------------------
// Launch
// ---------------------------------------------------------------------------
extern "C" void kernel_launch(void* const* d_in, const int* in_sizes, int n_in,
                              void* d_out, int out_size, void* d_ws, size_t ws_size,
                              hipStream_t stream)
{
    const float* x       = (const float*)d_in[0];
    const int*   mask    = (const int*)d_in[1];
    const float* ln_g    = (const float*)d_in[2];
    const float* ln_b    = (const float*)d_in[3];
    const float* w_qkv   = (const float*)d_in[4];
    const float* w_out   = (const float*)d_in[5];
    float*       out     = (float*)d_out;

    const size_t per_qkv  = (size_t)BATCH * HEADS * SEQ * DH;   // 4M f16
    const size_t xn_elems = (size_t)MTOK * DIMX;                // 4M f16
    _Float16* xn = (_Float16*)d_ws;
    _Float16* qb = xn + xn_elems;
    _Float16* kb = qb + per_qkv;
    _Float16* vb = kb + per_qkv;
    _Float16* ao = vb + per_qkv;

    ln_kernel<<<MTOK, 256, 0, stream>>>(x, ln_g, ln_b, xn);

    dim3 g1(MTOK / BM, (3 * INNER) / BN);
    gemm_kernel<3 * INNER, DIMX, 1><<<g1, 256, 0, stream>>>(
        xn, w_qkv, qb, kb, vb, nullptr);

    attn_kernel<<<BATCH * HEADS * (SEQ / 64), 128, 0, stream>>>(qb, kb, vb, mask, ao);

    dim3 g2(MTOK / BM, DIMX / BN);
    gemm_kernel<DIMX, DIMX, 0><<<g2, 256, 0, stream>>>(
        ao, w_out, nullptr, nullptr, nullptr, out);
}
